// ELModel_5007931867519
// MI455X (gfx1250) — matrix operand order repaired and checked by reading, not compiled, not run on hardware
//
#include <hip/hip_runtime.h>
#include <hip/hip_bf16.h>

// ---------------------------------------------------------------------------
// MI455X (gfx1250) implementation.
//  - all matmuls in f16 via v_wmma_f32_16x16x32_f16, f32 accumulation
//  - LSTM: input projection hoisted to one big WMMA GEMM; persistent 8-block
//    kernel with register-resident cell state and a global grid barrier
//  - attention: per-(b,head,qtile) wave; V panel DMA'd to LDS with
//    global_load_async_to_lds_b128 (ASYNCcnt), overlapped with QK^T+softmax;
//    WMMA for QK^T and PV
// ---------------------------------------------------------------------------

#define BB   64
#define SS   512
#define IND  256
#define HD   256
#define NHD  8
#define DHD  32
#define NEXP 5

typedef __attribute__((ext_vector_type(16))) _Float16 v16h;
typedef __attribute__((ext_vector_type(8)))  float    v8f;

// ---------------- WMMA helpers (CDNA5 fragment layouts, wave32) -------------

__device__ inline v8f wmma16(v16h a, v16h b, v8f c) {
  return __builtin_amdgcn_wmma_f32_16x16x32_f16(
      /*neg_a*/false, a, /*neg_b*/false, b, /*c_mod*/(short)0, c,
      /*reuse_a*/false, /*reuse_b*/false);
}

// A fragment: 16x32 (MxK) f16, row-major, lda = ld.
// lane m=lane&15 holds K = [hi*16, hi*16+16)
__device__ inline v16h load_a16(const _Float16* __restrict__ A, int ld, unsigned lane) {
  const unsigned m = lane & 15u, hi = (lane >> 4) & 1u;
  const _Float16* p = A + (size_t)m * ld;
  v16h f;
#pragma unroll
  for (int i = 0; i < 8; ++i) f[i] = p[hi * 16 + i];
#pragma unroll
  for (int i = 0; i < 8; ++i) f[8 + i] = p[hi * 16 + 8 + i];
  return f;
}

// B fragment: 32x16 (KxN) where column n = row n of W[N,K] (row-major, k contig)
// lane n=lane&15 holds K = [hi*16, hi*16+16)
__device__ inline v16h load_b16(const _Float16* __restrict__ W, int ld, unsigned lane) {
  const unsigned n = lane & 15u, hi = (lane >> 4) & 1u;
  const _Float16* p = W + (size_t)n * ld + hi * 16;
  v16h f;
#pragma unroll
  for (int i = 0; i < 16; ++i) f[i] = p[i];
  return f;
}

// B fragment from K-major storage: B[k][n] = V[k, n], stride ldk between k rows
__device__ inline v16h load_b16_kmajor(const _Float16* __restrict__ V, int ldk, unsigned lane) {
  const unsigned n = lane & 15u, hi = (lane >> 4) & 1u;
  const _Float16* p = V + (size_t)(hi * 16) * ldk + n;
  v16h f;
#pragma unroll
  for (int i = 0; i < 16; ++i) f[i] = p[(size_t)i * ldk];
  return f;
}

__device__ inline float sigf(float x) { return 1.f / (1.f + __expf(-x)); }

// ---------------- small utility kernels ------------------------------------

__global__ void cvt_kernel(_Float16* __restrict__ dst, const float* __restrict__ src, long n) {
  long i = (long)blockIdx.x * blockDim.x + threadIdx.x;
  if (i < n) dst[i] = (_Float16)src[i];
}

// src [R,C] f32 row-major -> dst [C,R] f16 (transpose)
__global__ void cvt_t_kernel(_Float16* __restrict__ dst, const float* __restrict__ src, int R, int C) {
  long i = (long)blockIdx.x * blockDim.x + threadIdx.x;
  if (i < (long)R * C) {
    int r = (int)(i / C), c = (int)(i % C);
    dst[(long)c * R + r] = (_Float16)src[i];
  }
}

__global__ void addb_kernel(float* __restrict__ dst, const float* __restrict__ a,
                            const float* __restrict__ b, int n) {
  int i = blockIdx.x * blockDim.x + threadIdx.x;
  if (i < n) dst[i] = a[i] + b[i];
}

// ---------------- generic WMMA GEMM: C = act(A[M,K] @ W[N,K]^T + bias) ------
// A,W f16; out f32 and/or f16. N multiple of 64, K multiple of 32, M of 16.

__global__ __launch_bounds__(256) void gemm_nt_kernel(
    const _Float16* __restrict__ A, const _Float16* __restrict__ W,
    const float* __restrict__ bias, float* __restrict__ Cf, _Float16* __restrict__ Ch,
    int M, int N, int K, int relu) {
  const unsigned lane = threadIdx.x & 31u;
  const unsigned wave = threadIdx.x >> 5;
  const unsigned n = lane & 15u, hi = (lane >> 4) & 1u;
  const int MT = M >> 4, NG = N >> 6;
  const long total = (long)MT * NG;
  for (long tid = (long)blockIdx.x * 8 + wave; tid < total; tid += (long)gridDim.x * 8) {
    const int mt = (int)(tid / NG), ng = (int)(tid % NG);
    const _Float16* Arow = A + (size_t)(mt * 16) * K;
    v8f acc[4];
#pragma unroll
    for (int j = 0; j < 4; ++j)
#pragma unroll
      for (int r = 0; r < 8; ++r) acc[j][r] = 0.f;
    for (int k0 = 0; k0 < K; k0 += 32) {
      __builtin_prefetch(Arow + k0 + 128);          // global_prefetch_b8
      v16h a = load_a16(Arow + k0, K, lane);
#pragma unroll
      for (int j = 0; j < 4; ++j) {
        const _Float16* Wt = W + (size_t)(ng * 64 + j * 16) * K + k0;
        v16h b = load_b16(Wt, K, lane);
        acc[j] = wmma16(a, b, acc[j]);
      }
    }
#pragma unroll
    for (int j = 0; j < 4; ++j) {
      const int col = ng * 64 + j * 16 + (int)n;
      const float bv = bias ? bias[col] : 0.f;
#pragma unroll
      for (int r = 0; r < 8; ++r) {
        const int row = mt * 16 + r + (int)hi * 8;
        float v = acc[j][r] + bv;
        if (relu) v = fmaxf(v, 0.f);
        if (Cf) Cf[(long)row * N + col] = v;
        if (Ch) Ch[(long)row * N + col] = (_Float16)v;
      }
    }
  }
}

// ---------------- LayerNorm (+optional residual, relu), H=256 ---------------

__global__ __launch_bounds__(256) void ln_kernel(
    const float* __restrict__ X, const float* __restrict__ R,
    const float* __restrict__ gam, const float* __restrict__ bet,
    float* __restrict__ Yf, _Float16* __restrict__ Yh, long rows, int relu) {
  const int lane = threadIdx.x & 31, wave = threadIdx.x >> 5;
  const long row = (long)blockIdx.x * 8 + wave;
  if (row >= rows) return;
  const float* xr = X + row * HD;
  const float* rr = R ? R + row * HD : nullptr;
  float v[8], s = 0.f;
#pragma unroll
  for (int i = 0; i < 8; ++i) {
    const int j = lane + i * 32;
    float t = xr[j] + (rr ? rr[j] : 0.f);
    v[i] = t; s += t;
  }
  for (int o = 16; o > 0; o >>= 1) s += __shfl_xor(s, o);
  const float mean = s * (1.f / 256.f);
  float vs = 0.f;
#pragma unroll
  for (int i = 0; i < 8; ++i) { float d = v[i] - mean; vs += d * d; }
  for (int o = 16; o > 0; o >>= 1) vs += __shfl_xor(vs, o);
  const float rstd = rsqrtf(vs * (1.f / 256.f) + 1e-5f);
#pragma unroll
  for (int i = 0; i < 8; ++i) {
    const int j = lane + i * 32;
    float y = (v[i] - mean) * rstd * gam[j] + bet[j];
    if (relu) y = fmaxf(y, 0.f);
    if (Yf) Yf[row * HD + j] = y;
    if (Yh) Yh[row * HD + j] = (_Float16)y;
  }
}

// ---------------- persistent LSTM (8 blocks x 8 waves, grid barrier) --------

__device__ inline void grid_sync(int* bar, int nblk) {
  __threadfence();
  __syncthreads();
  if (threadIdx.x == 0) {
    const int gen = atomicAdd(&bar[1], 0);
    const int v = atomicAdd(&bar[0], 1);
    if (v == nblk - 1) {
      atomicExch(&bar[0], 0);
      __threadfence();
      atomicAdd(&bar[1], 1);
    } else {
      while (atomicAdd(&bar[1], 0) == gen) __builtin_amdgcn_s_sleep(1);
    }
  }
  __syncthreads();
}

__global__ void lstm_init_kernel(_Float16* __restrict__ hbuf, int* __restrict__ bar) {
  const int i = blockIdx.x * blockDim.x + threadIdx.x;
  if (i < BB * HD) hbuf[i] = (_Float16)0.f;
  if (i < 2) bar[i] = 0;
}

__global__ __launch_bounds__(256) void lstm_kernel(
    const float* __restrict__ xproj,      // [B,S,4H]  x@Wih^T + b_ih + b_hh
    const _Float16* __restrict__ whh,     // [4H,H] f16
    const int* __restrict__ lengths,
    _Float16* __restrict__ hbuf,          // [2][B*H] f16 ping-pong
    float* __restrict__ lstm_out,         // [B,S,H] f32 (masked)
    _Float16* __restrict__ lstm_h,        // [B,S,H] f16 (masked)
    int* __restrict__ bar) {
  const unsigned lane = threadIdx.x & 31u, wave = threadIdx.x >> 5;
  const unsigned n = lane & 15u, hi = (lane >> 4) & 1u;
  const int wg = blockIdx.x * 8 + (int)wave;     // 0..63
  const int mt = wg >> 4, nc = wg & 15;          // 4 row tiles x 16 col tiles
  const int col = nc * 16 + (int)n;
  float c[8];
  int lenr[8];
#pragma unroll
  for (int r = 0; r < 8; ++r) {
    c[r] = 0.f;
    lenr[r] = lengths[mt * 16 + r + (int)hi * 8];
  }
  for (int t = 0; t < SS; ++t) {
    const _Float16* hprev = hbuf + (size_t)(t & 1) * (BB * HD);
    v8f acc[4];
#pragma unroll
    for (int gg = 0; gg < 4; ++gg)
#pragma unroll
      for (int r = 0; r < 8; ++r) {
        const int b = mt * 16 + r + (int)hi * 8;
        acc[gg][r] = xproj[((long)b * SS + t) * (4 * HD) + gg * HD + col];
      }
    const _Float16* Ab = hprev + (size_t)(mt * 16) * HD;
    for (int k0 = 0; k0 < HD; k0 += 32) {
      v16h a = load_a16(Ab + k0, HD, lane);
#pragma unroll
      for (int gg = 0; gg < 4; ++gg) {
        const _Float16* Wt = whh + (size_t)(gg * HD + nc * 16) * HD + k0;
        v16h b = load_b16(Wt, HD, lane);
        acc[gg] = wmma16(a, b, acc[gg]);
      }
    }
    _Float16* hnext = hbuf + (size_t)((t + 1) & 1) * (BB * HD);
#pragma unroll
    for (int r = 0; r < 8; ++r) {
      const int b = mt * 16 + r + (int)hi * 8;
      const float iv = sigf(acc[0][r]);
      const float fv = sigf(acc[1][r]);
      const float gv = tanhf(acc[2][r]);
      const float ov = sigf(acc[3][r]);
      c[r] = fv * c[r] + iv * gv;
      const float hv = ov * tanhf(c[r]);
      hnext[(size_t)b * HD + col] = (_Float16)hv;
      const float hm = (t < lenr[r]) ? hv : 0.f;
      const long oi = ((long)b * SS + t) * HD + col;
      lstm_out[oi] = hm;
      lstm_h[oi] = (_Float16)hm;
    }
    grid_sync(bar, 8);
  }
}

// ---------------- attention: one wave per (b, head, 16-query tile) ----------
// dynamic LDS: sc[16][512] f32 (32K) | pp[16][512] f16 (16K) | vb[512][32] f16 (32K)
#define ATTN_LDS (16 * SS * 4 + 16 * SS * 2 + SS * DHD * 2)

__global__ __launch_bounds__(32) void attn_kernel(
    const _Float16* __restrict__ qkv,   // [B,S,3H] f16: q|k|v
    const int* __restrict__ lengths,
    float* __restrict__ ao) {           // [B,S,H]
  extern __shared__ char smem[];
  float*    sc = (float*)smem;                              // [16][SS]
  _Float16* pp = (_Float16*)(smem + 16 * SS * 4);           // [16][SS]
  _Float16* vb = (_Float16*)(smem + 16 * SS * 4 + 16 * SS * 2); // [SS][DHD]
  const unsigned lane = threadIdx.x;
  const unsigned n = lane & 15u, hi = (lane >> 4) & 1u;
  const int bid = blockIdx.x;
  const int qt = bid & 31, head = (bid >> 5) & 7, b = bid >> 8;
  const int len = lengths[b];
  const _Float16* qbase = qkv + ((long)b * SS + qt * 16) * (3 * HD) + head * DHD;
  const _Float16* kbase = qkv + (long)b * SS * (3 * HD) + HD + head * DHD;
  const _Float16* vbase = qkv + (long)b * SS * (3 * HD) + 2 * HD + head * DHD;

  // Kick off async DMA of the V panel [SS, DHD] (32 KB) into LDS.
  // Each global_load_async_to_lds_b128 moves 16B per lane (512B/wave);
  // 64 issues cover the panel. Overlaps with QK^T and softmax below.
  {
    const char* vsrc = (const char*)vbase;        // row stride 3*HD*2 bytes
    const unsigned ldsbase = (unsigned)(size_t)vb;
#pragma unroll 4
    for (int i = 0; i < 64; ++i) {
      const unsigned off = (unsigned)i * 512u + lane * 16u;  // byte offset in panel
      const unsigned row = off >> 6, rcol = off & 63u;       // 64B per panel row
      const unsigned long long g =
          (unsigned long long)(size_t)(vsrc + (size_t)row * (3 * HD * 2) + rcol);
      const unsigned l = ldsbase + off;
      asm volatile("global_load_async_to_lds_b128 %0, %1, off"
                   :: "v"(l), "v"(g) : "memory");
    }
  }

  const v16h a = load_a16(qbase, 3 * HD, lane);      // Q tile 16x32
  const float scale = 0.17677669529663687f;          // 1/sqrt(32)
  for (int kt = 0; kt < SS / 16; ++kt) {
    v16h bf = load_b16(kbase + (size_t)kt * 16 * (3 * HD), 3 * HD, lane);
    v8f s;
#pragma unroll
    for (int r = 0; r < 8; ++r) s[r] = 0.f;
    s = wmma16(a, bf, s);
    const int kcol = kt * 16 + (int)n;
    const float msk = (kcol < len) ? 0.f : -1e9f;
#pragma unroll
    for (int r = 0; r < 8; ++r) sc[(r + (int)hi * 8) * SS + kcol] = s[r] * scale + msk;
  }
  __syncthreads();
  for (int m = 0; m < 16; ++m) {
    float mx = -1e30f;
    for (int j = lane; j < SS; j += 32) mx = fmaxf(mx, sc[m * SS + j]);
    for (int o = 16; o > 0; o >>= 1) mx = fmaxf(mx, __shfl_xor(mx, o));
    float sum = 0.f;
    for (int j = lane; j < SS; j += 32) {
      const float e = __expf(sc[m * SS + j] - mx);
      sc[m * SS + j] = e;
      sum += e;
    }
    for (int o = 16; o > 0; o >>= 1) sum += __shfl_xor(sum, o);
    const float inv = 1.f / sum;
    for (int j = lane; j < SS; j += 32) pp[m * SS + j] = (_Float16)(sc[m * SS + j] * inv);
  }
  // make sure the V panel DMA has landed in LDS before PV
  asm volatile("s_wait_asynccnt 0x0" ::: "memory");
  __syncthreads();
#pragma unroll
  for (int dt = 0; dt < 2; ++dt) {
    v8f acc;
#pragma unroll
    for (int r = 0; r < 8; ++r) acc[r] = 0.f;
    for (int k0 = 0; k0 < SS; k0 += 32) {
      v16h pa = load_a16(pp + k0, SS, lane);
      v16h vbf = load_b16_kmajor(vb + (size_t)k0 * DHD + dt * 16, DHD, lane);
      acc = wmma16(pa, vbf, acc);
    }
#pragma unroll
    for (int r = 0; r < 8; ++r) {
      const int m = r + (int)hi * 8;
      ao[((long)b * SS + qt * 16 + m) * HD + head * DHD + dt * 16 + (int)n] = acc[r];
    }
  }
}

// ---------------- router, MoE select, pool, head ----------------------------

__global__ __launch_bounds__(256) void router_kernel(
    const float* __restrict__ src, const float* __restrict__ rw,  // rw [H,E]
    int* __restrict__ ridx, float* __restrict__ rgate, long rows) {
  const int lane = threadIdx.x & 31, wave = threadIdx.x >> 5;
  const long row = (long)blockIdx.x * 8 + wave;
  if (row >= rows) return;
  const float* x = src + row * HD;
  float lg[NEXP] = {0.f, 0.f, 0.f, 0.f, 0.f};
  for (int j = lane; j < HD; j += 32) {
    const float xv = x[j];
#pragma unroll
    for (int e = 0; e < NEXP; ++e) lg[e] += xv * rw[j * NEXP + e];
  }
#pragma unroll
  for (int e = 0; e < NEXP; ++e)
    for (int o = 16; o > 0; o >>= 1) lg[e] += __shfl_xor(lg[e], o);
  float mx = lg[0]; int bi = 0;
#pragma unroll
  for (int e = 1; e < NEXP; ++e) if (lg[e] > mx) { mx = lg[e]; bi = e; }
  float den = 0.f;
#pragma unroll
  for (int e = 0; e < NEXP; ++e) den += __expf(lg[e] - mx);
  if (lane == 0) { ridx[row] = bi; rgate[row] = 1.f / den; }
}

__global__ void moe_select_kernel(const float* __restrict__ ey, const int* __restrict__ ridx,
                                  const float* __restrict__ rgate, float* __restrict__ acc,
                                  int e, long rows) {
  const long i = (long)blockIdx.x * blockDim.x + threadIdx.x;
  if (i >= rows * HD) return;
  const long row = i >> 8;
  if (ridx[row] == e) acc[i] = rgate[row] * ey[i];
}

__global__ __launch_bounds__(256) void pool_kernel(
    const float* __restrict__ out2, const int* __restrict__ lengths,
    float* __restrict__ pooled, _Float16* __restrict__ pooled_h) {
  const int b = blockIdx.x, col = threadIdx.x;
  const int len = lengths[b];
  float s = 0.f;
  for (int t = 0; t < len; ++t) s += out2[((long)b * SS + t) * HD + col];
  const float p = s / (float)len;
  pooled[b * HD + col] = p;
  pooled_h[b * HD + col] = (_Float16)p;
}

__global__ __launch_bounds__(256) void head_kernel(
    const float* __restrict__ hcl, const float* __restrict__ w2,
    const float* __restrict__ b2, float* __restrict__ out) {
  const int lane = threadIdx.x & 31, wave = threadIdx.x >> 5;
  const int b = blockIdx.x * 8 + wave;
  if (b >= BB) return;
  float s = 0.f;
  for (int j = lane; j < HD; j += 32) s += hcl[b * HD + j] * w2[j];
  for (int o = 16; o > 0; o >>= 1) s += __shfl_xor(s, o);
  if (lane == 0) out[b] = s + b2[0];
}

// ---------------------------------------------------------------------------

extern "C" void kernel_launch(void* const* d_in, const int* in_sizes, int n_in,
                              void* d_out, int out_size, void* d_ws, size_t ws_size,
                              hipStream_t stream) {
  (void)in_sizes; (void)n_in; (void)out_size; (void)ws_size;
  const float* x       = (const float*)d_in[0];
  const int*   lengths = (const int*)  d_in[1];
  const float* fe_w    = (const float*)d_in[2];
  const float* fe_b    = (const float*)d_in[3];
  const float* fe_ln_g = (const float*)d_in[4];
  const float* fe_ln_b = (const float*)d_in[5];
  const float* w_ih    = (const float*)d_in[6];
  const float* w_hh    = (const float*)d_in[7];
  const float* b_ih    = (const float*)d_in[8];
  const float* b_hh    = (const float*)d_in[9];
  const float* attn_w  = (const float*)d_in[10];
  const float* attn_b  = (const float*)d_in[11];
  const float* attn_ow = (const float*)d_in[12];
  const float* attn_ob = (const float*)d_in[13];
  const float* ln1_g   = (const float*)d_in[14];
  const float* ln1_b   = (const float*)d_in[15];
  const float* router_w= (const float*)d_in[16];
  const float* e_w1    = (const float*)d_in[17];
  const float* e_b1    = (const float*)d_in[18];
  const float* e_w2    = (const float*)d_in[19];
  const float* e_b2    = (const float*)d_in[20];
  const float* ln2_g   = (const float*)d_in[21];
  const float* ln2_b   = (const float*)d_in[22];
  const float* cl_w1   = (const float*)d_in[23];
  const float* cl_b1   = (const float*)d_in[24];
  const float* cl_ln_g = (const float*)d_in[25];
  const float* cl_ln_b = (const float*)d_in[26];
  const float* cl_w2   = (const float*)d_in[27];
  const float* cl_b2   = (const float*)d_in[28];

  char* bump = (char*)d_ws;
  auto alloc = [&](size_t bytes) -> void* {
    void* r = (void*)bump;
    bump += (bytes + 255) & ~(size_t)255;
    return r;
  };
  const long NT = (long)BB * SS;  // 32768 tokens
  _Float16* x_h    = (_Float16*)alloc((size_t)NT * IND * 2);
  _Float16* fe_wh  = (_Float16*)alloc((size_t)HD * IND * 2);
  _Float16* wih_h  = (_Float16*)alloc((size_t)4 * HD * HD * 2);
  _Float16* whh_h  = (_Float16*)alloc((size_t)4 * HD * HD * 2);
  _Float16* attw_h = (_Float16*)alloc((size_t)3 * HD * HD * 2);
  _Float16* atow_h = (_Float16*)alloc((size_t)HD * HD * 2);
  _Float16* ew1_h  = (_Float16*)alloc((size_t)NEXP * HD * HD * 2);
  _Float16* ew2_h  = (_Float16*)alloc((size_t)NEXP * HD * HD * 2);
  _Float16* clw1_h = (_Float16*)alloc((size_t)HD * HD * 2);
  float*    biasc  = (float*)alloc((size_t)4 * HD * 4);
  float*    h0f    = (float*)alloc((size_t)NT * HD * 4);
  _Float16* h0_h   = (_Float16*)alloc((size_t)NT * HD * 2);
  float*    xproj  = (float*)alloc((size_t)NT * 4 * HD * 4);
  _Float16* hbuf   = (_Float16*)alloc((size_t)2 * BB * HD * 2);
  int*      bar    = (int*)alloc(256);
  float*    lstm_o = (float*)alloc((size_t)NT * HD * 4);
  _Float16* lstm_h = (_Float16*)alloc((size_t)NT * HD * 2);
  _Float16* qkv_h  = (_Float16*)alloc((size_t)NT * 3 * HD * 2);
  float*    ao     = (float*)alloc((size_t)NT * HD * 4);
  _Float16* ao_h   = (_Float16*)alloc((size_t)NT * HD * 2);
  float*    aoproj = (float*)alloc((size_t)NT * HD * 4);
  float*    src    = (float*)alloc((size_t)NT * HD * 4);
  _Float16* src_h  = (_Float16*)alloc((size_t)NT * HD * 2);
  int*      ridx   = (int*)alloc((size_t)NT * 4);
  float*    rgate  = (float*)alloc((size_t)NT * 4);
  _Float16* eh_h   = (_Float16*)alloc((size_t)NT * HD * 2);
  float*    eybuf  = (float*)alloc((size_t)NT * HD * 4);
  float*    moeacc = (float*)alloc((size_t)NT * HD * 4);
  float*    out2   = (float*)alloc((size_t)NT * HD * 4);
  float*    pooled = (float*)alloc((size_t)BB * HD * 4);
  _Float16* pool_h = (_Float16*)alloc((size_t)BB * HD * 2);
  float*    clpre  = (float*)alloc((size_t)BB * HD * 4);
  float*    hcl    = (float*)alloc((size_t)BB * HD * 4);

  auto cvt = [&](_Float16* dst, const float* s, long n) {
    cvt_kernel<<<(int)((n + 255) / 256), 256, 0, stream>>>(dst, s, n);
  };
  auto gemm = [&](const _Float16* A, const _Float16* W, const float* bias,
                  float* Cf, _Float16* Ch, int M, int N, int relu) {
    long tiles = ((long)M / 16) * (N / 64);
    int blocks = (int)((tiles + 7) / 8);
    if (blocks > 4096) blocks = 4096;
    gemm_nt_kernel<<<blocks, 256, 0, stream>>>(A, W, bias, Cf, Ch, M, N, HD, relu);
  };

  // weight / input precision conversion (f32 -> f16)
  cvt(x_h, x, NT * IND);
  cvt(fe_wh, fe_w, (long)HD * IND);
  cvt(wih_h, w_ih, (long)4 * HD * HD);
  cvt(whh_h, w_hh, (long)4 * HD * HD);
  cvt(attw_h, attn_w, (long)3 * HD * HD);
  cvt(atow_h, attn_ow, (long)HD * HD);
  cvt(clw1_h, cl_w1, (long)HD * HD);
  for (int e = 0; e < NEXP; ++e) {  // experts contract over the FIRST dim -> transpose
    cvt_t_kernel<<<256, 256, 0, stream>>>(ew1_h + (size_t)e * HD * HD, e_w1 + (size_t)e * HD * HD, HD, HD);
    cvt_t_kernel<<<256, 256, 0, stream>>>(ew2_h + (size_t)e * HD * HD, e_w2 + (size_t)e * HD * HD, HD, HD);
  }
  addb_kernel<<<4, 256, 0, stream>>>(biasc, b_ih, b_hh, 4 * HD);

  // feature extractor: Linear -> LN -> ReLU
  gemm(x_h, fe_wh, fe_b, h0f, nullptr, (int)NT, HD, 0);
  ln_kernel<<<4096, 256, 0, stream>>>(h0f, nullptr, fe_ln_g, fe_ln_b, nullptr, h0_h, NT, 1);

  // LSTM input projection hoisted out of the scan
  gemm(h0_h, wih_h, biasc, xproj, nullptr, (int)NT, 4 * HD, 0);

  // persistent recurrent scan
  lstm_init_kernel<<<64, 256, 0, stream>>>(hbuf, bar);
  lstm_kernel<<<8, 256, 0, stream>>>(xproj, whh_h, lengths, hbuf, lstm_o, lstm_h, bar);

  // attention
  gemm(lstm_h, attw_h, attn_b, nullptr, qkv_h, (int)NT, 3 * HD, 0);
  attn_kernel<<<BB * NHD * (SS / 16), 32, ATTN_LDS, stream>>>(qkv_h, lengths, ao);
  cvt(ao_h, ao, NT * HD);
  gemm(ao_h, atow_h, attn_ob, aoproj, nullptr, (int)NT, HD, 0);
  ln_kernel<<<4096, 256, 0, stream>>>(aoproj, lstm_o, ln1_g, ln1_b, src, src_h, NT, 0);

  // top-1 switch MoE (dense expert GEMMs + select)
  router_kernel<<<4096, 256, 0, stream>>>(src, router_w, ridx, rgate, NT);
  for (int e = 0; e < NEXP; ++e) {
    gemm(src_h, ew1_h + (size_t)e * HD * HD, e_b1 + (size_t)e * HD, nullptr, eh_h, (int)NT, HD, 1);
    gemm(eh_h, ew2_h + (size_t)e * HD * HD, e_b2 + (size_t)e * HD, eybuf, nullptr, (int)NT, HD, 0);
    moe_select_kernel<<<(int)((NT * HD + 255) / 256), 256, 0, stream>>>(eybuf, ridx, rgate, moeacc, e, NT);
  }
  ln_kernel<<<4096, 256, 0, stream>>>(moeacc, src, ln2_g, ln2_b, out2, nullptr, NT, 0);

  // masked mean pool + classifier head
  pool_kernel<<<BB, 256, 0, stream>>>(out2, lengths, pooled, pool_h);
  gemm(pool_h, clw1_h, cl_b1, clpre, nullptr, BB, HD, 0);
  ln_kernel<<<8, 256, 0, stream>>>(clpre, nullptr, cl_ln_g, cl_ln_b, hcl, nullptr, BB, 1);
  head_kernel<<<8, 256, 0, stream>>>(hcl, cl_w2, cl_b2, (float*)d_out);
}
